// HardNegativeWeightedPConLoss_63866163691790
// MI455X (gfx1250) — matrix operand order repaired and checked by reading, not compiled, not run on hardware
//
#include <hip/hip_runtime.h>
#include <hip/hip_bf16.h>

// ---------------------------------------------------------------------------
// Types for CDNA5 WMMA (wave32): v_wmma_f32_16x16x32_bf16
// ---------------------------------------------------------------------------
typedef __attribute__((ext_vector_type(16))) __bf16 v16bf;
typedef __attribute__((ext_vector_type(8)))  __bf16 v8bf;    // 16 bytes -> ds_load_b128
typedef __attribute__((ext_vector_type(8)))  float  v8f;

#define NNODES 4096
#define HDIM   256
#define TAU_INV 10.0f
#define WTEMP_INV (1.0f/0.7f)

// ---------------------------------------------------------------------------
// 0) zero accumulators
// ---------------------------------------------------------------------------
__global__ void zero_kernel(float* p, int n) {
    int i = blockIdx.x * 256 + threadIdx.x;
    if (i < n) p[i] = 0.f;
}

// ---------------------------------------------------------------------------
// 1) L2-normalize each embedding row -> bf16
// ---------------------------------------------------------------------------
__global__ __launch_bounds__(256)
void prep_norm_kernel(const float* __restrict__ P, const float* __restrict__ M,
                      const float* __restrict__ F,
                      __bf16* __restrict__ Pn, __bf16* __restrict__ Mn,
                      __bf16* __restrict__ Fn) {
    __shared__ float red[256];
    __shared__ float sinv;
    const int row = blockIdx.x, t = threadIdx.x;
    const float* src[3] = {P, M, F};
    __bf16* dst[3] = {Pn, Mn, Fn};
    for (int e = 0; e < 3; ++e) {
        float x = src[e][(size_t)row * HDIM + t];
        red[t] = x * x;
        __syncthreads();
        for (int s = 128; s; s >>= 1) { if (t < s) red[t] += red[t + s]; __syncthreads(); }
        if (t == 0) sinv = 1.f / fmaxf(sqrtf(red[0]), 1e-12f);
        __syncthreads();
        dst[e][(size_t)row * HDIM + t] = (__bf16)(x * sinv);
        __syncthreads();
    }
}

// ---------------------------------------------------------------------------
// 2) adjacency row sums (degrees)
// ---------------------------------------------------------------------------
__global__ __launch_bounds__(256)
void rowsum_kernel(const float* __restrict__ a0, const float* __restrict__ a1,
                   const float* __restrict__ a2,
                   float* __restrict__ d0, float* __restrict__ d1, float* __restrict__ d2) {
    __shared__ float red[256];
    const float* A = (blockIdx.y == 0) ? a0 : (blockIdx.y == 1) ? a1 : a2;
    float*       D = (blockIdx.y == 0) ? d0 : (blockIdx.y == 1) ? d1 : d2;
    const int row = blockIdx.x, t = threadIdx.x;
    float s = 0.f;
    for (int c = t; c < NNODES; c += 256) s += A[(size_t)row * NNODES + c];
    red[t] = s;
    __syncthreads();
    for (int q = 128; q; q >>= 1) { if (t < q) red[t] += red[t + q]; __syncthreads(); }
    if (t == 0) D[row] = red[0];
}

// ---------------------------------------------------------------------------
// Staging helpers (register-staged f32 -> bf16 tile store)
// ---------------------------------------------------------------------------
__device__ __forceinline__ void stage_load_f32(const float* __restrict__ A,
                                               const float* __restrict__ B,
                                               int K, int ldb, int rowBase, int colBase,
                                               int k0, int tid, float4* ra, float4* rb) {
    #pragma unroll
    for (int i = 0; i < 4; ++i) {
        int idx = tid + i * 256, r = idx >> 4, c4 = idx & 15;
        ra[i] = *(const float4*)&A[(size_t)(rowBase + r) * K + k0 + c4 * 4];
    }
    #pragma unroll
    for (int i = 0; i < 4; ++i) {
        int idx = tid + i * 256, k = idx >> 4, n4 = idx & 15;
        rb[i] = *(const float4*)&B[(size_t)(k0 + k) * ldb + colBase + n4 * 4];
    }
}

__device__ __forceinline__ void stage_store_f32(__bf16 (*as)[72], __bf16 (*bs)[72],
                                                const float4* ra, const float4* rb, int tid) {
    #pragma unroll
    for (int i = 0; i < 4; ++i) {
        int idx = tid + i * 256, r = idx >> 4, c4 = idx & 15;
        __bf16* d = &as[r][c4 * 4];
        d[0] = (__bf16)ra[i].x; d[1] = (__bf16)ra[i].y;
        d[2] = (__bf16)ra[i].z; d[3] = (__bf16)ra[i].w;
    }
    #pragma unroll
    for (int i = 0; i < 4; ++i) {
        int idx = tid + i * 256, k = idx >> 4, n4 = idx & 15;   // transpose: Bs[n][k]
        bs[n4 * 4 + 0][k] = (__bf16)rb[i].x; bs[n4 * 4 + 1][k] = (__bf16)rb[i].y;
        bs[n4 * 4 + 2][k] = (__bf16)rb[i].z; bs[n4 * 4 + 3][k] = (__bf16)rb[i].w;
    }
}

// ---------------------------------------------------------------------------
// 3) Generic bf16 WMMA GEMM: C[M x N] = A(f32,MxK) @ B(f32,KxN).
//    Double-buffered LDS; next tile's global loads issue before the wmma loop.
//    Block tile 64x64, 8 waves, wave tile 16x32 (two 16x16x32 accumulators).
//    mode 0: out = acc / max(deg[row],1)  (aggregation -> feat at colOff)
//    mode 1: out = relu(acc + bias[col])  (MLP layer 1)
// ---------------------------------------------------------------------------
__global__ __launch_bounds__(256)
void gemm_bf16_kernel(const float* __restrict__ A, const float* __restrict__ B,
                      int K, int ldb,
                      float* __restrict__ out, int ldOut, int colOff,
                      const float* __restrict__ deg, const float* __restrict__ bias,
                      int mode) {
    __shared__ __bf16 As[2][64][72];   // row-major [m][k]
    __shared__ __bf16 Bs[2][64][72];   // transposed [n][k]
    const int tid = threadIdx.x;
    const int wave = tid >> 5, lane = tid & 31;
    const int wm = wave >> 1, wn = wave & 1;
    const int l15 = lane & 15, lh = lane >> 4;
    const int rowBase = blockIdx.y * 64;
    const int colBase = blockIdx.x * 64;
    v8f acc0 = {}, acc1 = {};
    float4 ra[4], rb[4];

    stage_load_f32(A, B, K, ldb, rowBase, colBase, 0, tid, ra, rb);
    stage_store_f32(As[0], Bs[0], ra, rb, tid);

    int buf = 0;
    for (int k0 = 0; k0 < K; k0 += 64) {
        __syncthreads();
        const bool more = (k0 + 64 < K);
        if (more) {   // issue next tile's loads; wmma below covers their latency
            stage_load_f32(A, B, K, ldb, rowBase, colBase, k0 + 64, tid, ra, rb);
            if (k0 + 128 < K) {   // global_prefetch_b8 two tiles ahead
                __builtin_prefetch(&A[(size_t)(rowBase + (tid >> 2)) * K + k0 + 128], 0, 1);
                __builtin_prefetch(&B[(size_t)(k0 + 128 + (tid >> 2)) * ldb + colBase], 0, 1);
            }
        }
        #pragma unroll
        for (int kk = 0; kk < 64; kk += 32) {
            const int ar = wm * 16 + l15;
            v16bf a, b0, b1;
            {   // A 16x32 fragment: lane holds one row M, contiguous 16 K
                const v8bf alo = *(const v8bf*)&As[buf][ar][kk + lh * 16];
                const v8bf ahi = *(const v8bf*)&As[buf][ar][kk + lh * 16 + 8];
                #pragma unroll
                for (int i = 0; i < 8; ++i) { a[i] = alo[i]; a[i + 8] = ahi[i]; }
            }
            {   // B 32x16 fragment: lane holds one column N, contiguous 16 K
                const int bc = wn * 32 + l15;
                const v8bf blo = *(const v8bf*)&Bs[buf][bc][kk + lh * 16];
                const v8bf bhi = *(const v8bf*)&Bs[buf][bc][kk + lh * 16 + 8];
                #pragma unroll
                for (int i = 0; i < 8; ++i) { b0[i] = blo[i]; b0[i + 8] = bhi[i]; }
            }
            {
                const int bc = wn * 32 + 16 + l15;
                const v8bf blo = *(const v8bf*)&Bs[buf][bc][kk + lh * 16];
                const v8bf bhi = *(const v8bf*)&Bs[buf][bc][kk + lh * 16 + 8];
                #pragma unroll
                for (int i = 0; i < 8; ++i) { b1[i] = blo[i]; b1[i + 8] = bhi[i]; }
            }
            acc0 = __builtin_amdgcn_wmma_f32_16x16x32_bf16(false, a, false, b0, (short)0, acc0, false, false);
            acc1 = __builtin_amdgcn_wmma_f32_16x16x32_bf16(false, a, false, b1, (short)0, acc1, false, false);
        }
        if (more) stage_store_f32(As[buf ^ 1], Bs[buf ^ 1], ra, rb, tid);
        buf ^= 1;
    }
    // epilogue: C layout VGPR r -> M = r + 8*(lane>=16), N = lane&15
    #pragma unroll
    for (int r = 0; r < 8; ++r) {
        const int gRow = rowBase + wm * 16 + r + lh * 8;
        #pragma unroll
        for (int s = 0; s < 2; ++s) {
            const int cc = colBase + wn * 32 + s * 16 + l15;
            float v = s ? acc1[r] : acc0[r];
            if (mode == 0) v = v / fmaxf(deg[gRow], 1.f);
            else           v = fmaxf(v + bias[cc], 0.f);
            out[(size_t)gRow * ldOut + colOff + cc] = v;
        }
    }
}

// ---------------------------------------------------------------------------
// 4) MLP layer 2 + double softmax -> weights [N,3] (written into d_out+1)
// ---------------------------------------------------------------------------
__global__ __launch_bounds__(256)
void mlp2_weights_kernel(const float* __restrict__ h, const float* __restrict__ W2,
                         const float* __restrict__ b2, float* __restrict__ Wout) {
    const int wave = threadIdx.x >> 5, lane = threadIdx.x & 31;
    const int row = blockIdx.x * 8 + wave;
    float z0 = 0.f, z1 = 0.f, z2 = 0.f;
    for (int k = lane; k < HDIM; k += 32) {
        float hv = h[(size_t)row * HDIM + k];
        z0 += hv * W2[k * 3 + 0];
        z1 += hv * W2[k * 3 + 1];
        z2 += hv * W2[k * 3 + 2];
    }
    for (int o = 16; o; o >>= 1) {
        z0 += __shfl_down(z0, o); z1 += __shfl_down(z1, o); z2 += __shfl_down(z2, o);
    }
    if (lane == 0) {
        z0 += b2[0]; z1 += b2[1]; z2 += b2[2];
        float m = fmaxf(z0, fmaxf(z1, z2));
        float e0 = __expf(z0 - m), e1 = __expf(z1 - m), e2 = __expf(z2 - m);
        float is = 1.f / (e0 + e1 + e2);
        float r0 = e0 * is, r1 = e1 * is, r2 = e2 * is;
        float m2 = fmaxf(r0, fmaxf(r1, r2)) * WTEMP_INV;
        float f0 = __expf(r0 * WTEMP_INV - m2), f1 = __expf(r1 * WTEMP_INV - m2),
              f2 = __expf(r2 * WTEMP_INV - m2);
        float is2 = 1.f / (f0 + f1 + f2);
        Wout[row * 3 + 0] = f0 * is2;
        Wout[row * 3 + 1] = f1 * is2;
        Wout[row * 3 + 2] = f2 * is2;
    }
}

// ---------------------------------------------------------------------------
// 5) Similarity GEMM (bf16 WMMA, double-buffered) with fused epilogue:
//    s = (Pn . Xn)/TAU ; e = exp(s)
//    pos[row] += e where adj!=0  (shfl + LDS reduce, one f32 atomic per row)
//    Sout[row,col] = adj!=0 ? -inf(h) : (half)s   (only for PP / PF)
// ---------------------------------------------------------------------------
__global__ __launch_bounds__(256)
void sim_pos_kernel(const __bf16* __restrict__ An, const __bf16* __restrict__ Bn,
                    const float* __restrict__ adj, float* __restrict__ pos,
                    _Float16* __restrict__ Sout) {
    __shared__ __bf16 As[2][64][72];
    __shared__ __bf16 Bs[2][64][72];
    __shared__ float posLDS[64];
    const int tid = threadIdx.x;
    const int wave = tid >> 5, lane = tid & 31;
    const int wm = wave >> 1, wn = wave & 1;
    const int l15 = lane & 15, lh = lane >> 4;
    const int rowBase = blockIdx.y * 64;
    const int colBase = blockIdx.x * 64;
    if (tid < 64) posLDS[tid] = 0.f;
    v8f acc0 = {}, acc1 = {};
    v8bf ra[2], rb[2];

    const int c0 = tid, c1 = tid + 256;            // 512 chunks of 8 bf16 per tile
    const int r0 = c0 >> 3, k80 = c0 & 7;
    const int r1 = c1 >> 3, k81 = c1 & 7;

    ra[0] = *(const v8bf*)&An[(size_t)(rowBase + r0) * HDIM + k80 * 8];
    ra[1] = *(const v8bf*)&An[(size_t)(rowBase + r1) * HDIM + k81 * 8];
    rb[0] = *(const v8bf*)&Bn[(size_t)(colBase + r0) * HDIM + k80 * 8];
    rb[1] = *(const v8bf*)&Bn[(size_t)(colBase + r1) * HDIM + k81 * 8];
    *(v8bf*)&As[0][r0][k80 * 8] = ra[0];
    *(v8bf*)&As[0][r1][k81 * 8] = ra[1];
    *(v8bf*)&Bs[0][r0][k80 * 8] = rb[0];
    *(v8bf*)&Bs[0][r1][k81 * 8] = rb[1];

    int buf = 0;
    for (int k0 = 0; k0 < HDIM; k0 += 64) {
        __syncthreads();
        const bool more = (k0 + 64 < HDIM);
        if (more) {
            ra[0] = *(const v8bf*)&An[(size_t)(rowBase + r0) * HDIM + k0 + 64 + k80 * 8];
            ra[1] = *(const v8bf*)&An[(size_t)(rowBase + r1) * HDIM + k0 + 64 + k81 * 8];
            rb[0] = *(const v8bf*)&Bn[(size_t)(colBase + r0) * HDIM + k0 + 64 + k80 * 8];
            rb[1] = *(const v8bf*)&Bn[(size_t)(colBase + r1) * HDIM + k0 + 64 + k81 * 8];
        }
        #pragma unroll
        for (int kk = 0; kk < 64; kk += 32) {
            const int ar = wm * 16 + l15;
            v16bf a, b0, b1;
            {
                const v8bf alo = *(const v8bf*)&As[buf][ar][kk + lh * 16];
                const v8bf ahi = *(const v8bf*)&As[buf][ar][kk + lh * 16 + 8];
                #pragma unroll
                for (int i = 0; i < 8; ++i) { a[i] = alo[i]; a[i + 8] = ahi[i]; }
            }
            {
                const int bc = wn * 32 + l15;
                const v8bf blo = *(const v8bf*)&Bs[buf][bc][kk + lh * 16];
                const v8bf bhi = *(const v8bf*)&Bs[buf][bc][kk + lh * 16 + 8];
                #pragma unroll
                for (int i = 0; i < 8; ++i) { b0[i] = blo[i]; b0[i + 8] = bhi[i]; }
            }
            {
                const int bc = wn * 32 + 16 + l15;
                const v8bf blo = *(const v8bf*)&Bs[buf][bc][kk + lh * 16];
                const v8bf bhi = *(const v8bf*)&Bs[buf][bc][kk + lh * 16 + 8];
                #pragma unroll
                for (int i = 0; i < 8; ++i) { b1[i] = blo[i]; b1[i + 8] = bhi[i]; }
            }
            acc0 = __builtin_amdgcn_wmma_f32_16x16x32_bf16(false, a, false, b0, (short)0, acc0, false, false);
            acc1 = __builtin_amdgcn_wmma_f32_16x16x32_bf16(false, a, false, b1, (short)0, acc1, false, false);
        }
        if (more) {
            *(v8bf*)&As[buf ^ 1][r0][k80 * 8] = ra[0];
            *(v8bf*)&As[buf ^ 1][r1][k81 * 8] = ra[1];
            *(v8bf*)&Bs[buf ^ 1][r0][k80 * 8] = rb[0];
            *(v8bf*)&Bs[buf ^ 1][r1][k81 * 8] = rb[1];
        }
        buf ^= 1;
    }

    const _Float16 NEGINF_H = __builtin_bit_cast(_Float16, (unsigned short)0xFC00u);
    #pragma unroll
    for (int r = 0; r < 8; ++r) {
        const int gRow = rowBase + wm * 16 + r + lh * 8;
        float rowp = 0.f;
        #pragma unroll
        for (int s = 0; s < 2; ++s) {
            const int gCol = colBase + wn * 32 + s * 16 + l15;
            const float sv = (s ? acc1[r] : acc0[r]) * TAU_INV;
            const float a  = adj[(size_t)gRow * NNODES + gCol];
            const float e  = __expf(sv);
            rowp += (a != 0.f) ? e : 0.f;
            if (Sout) {
                _Float16 hv = (a != 0.f) ? NEGINF_H : (_Float16)sv;
                Sout[(size_t)gRow * NNODES + gCol] = hv;
            }
        }
        for (int o = 8; o; o >>= 1) rowp += __shfl_down(rowp, o, 16);
        if (l15 == 0) atomicAdd(&posLDS[wm * 16 + r + lh * 8], rowp);
    }
    __syncthreads();
    if (tid < 64) atomicAdd(&pos[rowBase + tid], posLDS[tid]);
}

// ---------------------------------------------------------------------------
// 6) Per-row hard-negative selection: radix-select on fp16 bit keys.
//    negsum[row] = sum_neg e + sum_{top-k neg} e   (hard weight 2.0)
// ---------------------------------------------------------------------------
__global__ __launch_bounds__(256)
void neg_select_kernel(const _Float16* __restrict__ S, const float* __restrict__ deg,
                       float* __restrict__ negsum) {
    __shared__ unsigned hist[256];
    __shared__ float redA[256], redH[256];
    __shared__ int sh_hi, sh_cab, sh_key, sh_need;
    const int row = blockIdx.x, t = threadIdx.x;
    const _Float16* Srow = S + (size_t)row * NNODES;

    float vals[16];
    unsigned short keys[16];
    #pragma unroll
    for (int j = 0; j < 16; ++j) {
        _Float16 hv = Srow[t + j * 256];
        unsigned short u = __builtin_bit_cast(unsigned short, hv);
        keys[j] = (u & 0x8000u) ? (unsigned short)(~u) : (unsigned short)(u | 0x8000u);
        vals[j] = (float)hv;
    }
    const int num_neg = NNODES - (int)deg[row];
    int k = (int)((float)num_neg * 0.3f);
    if (k < 1) k = 1;

    hist[t] = 0;
    __syncthreads();
    #pragma unroll
    for (int j = 0; j < 16; ++j) atomicAdd(&hist[keys[j] >> 8], 1u);
    __syncthreads();
    if (t == 0) {
        int cum = 0, b = 255;
        for (; b >= 0; --b) { int hh = (int)hist[b]; if (cum + hh >= k) break; cum += hh; }
        sh_hi = b; sh_cab = cum;
    }
    __syncthreads();
    const int hi = sh_hi, cab = sh_cab;
    hist[t] = 0;
    __syncthreads();
    #pragma unroll
    for (int j = 0; j < 16; ++j)
        if ((int)(keys[j] >> 8) == hi) atomicAdd(&hist[keys[j] & 0xFF], 1u);
    __syncthreads();
    if (t == 0) {
        int cum = cab, b = 255;
        for (; b >= 0; --b) { int hh = (int)hist[b]; if (cum + hh >= k) break; cum += hh; }
        sh_key = (hi << 8) | b;
        sh_need = k - cum;
    }
    __syncthreads();
    const unsigned tkey = (unsigned)sh_key;

    float sAll = 0.f, sHard = 0.f;
    #pragma unroll
    for (int j = 0; j < 16; ++j) {
        float e = __expf(vals[j]);   // exp(-inf)=0 kills masked positives
        sAll += e;
        if ((unsigned)keys[j] > tkey) sHard += e;
    }
    redA[t] = sAll; redH[t] = sHard;
    __syncthreads();
    for (int s = 128; s; s >>= 1) {
        if (t < s) { redA[t] += redA[t + s]; redH[t] += redH[t + s]; }
        __syncthreads();
    }
    if (t == 0) {
        unsigned short u = (tkey & 0x8000u) ? (unsigned short)(tkey & 0x7FFFu)
                                            : (unsigned short)(~tkey);
        float tv = (float)__builtin_bit_cast(_Float16, u);
        negsum[row] = redA[0] + redH[0] + (float)sh_need * __expf(tv);
    }
}

// ---------------------------------------------------------------------------
// 7) Per-row loss + deterministic two-stage mean
// ---------------------------------------------------------------------------
__global__ __launch_bounds__(256)
void final_row_kernel(const float* __restrict__ W,
                      const float* __restrict__ pPM, const float* __restrict__ pPP,
                      const float* __restrict__ pPF,
                      const float* __restrict__ nPP, const float* __restrict__ nPF,
                      const float* __restrict__ dPM, const float* __restrict__ dPP,
                      const float* __restrict__ dPF,
                      float* __restrict__ partials) {
    __shared__ float red[256];
    const int row = blockIdx.x * 256 + threadIdx.x;
    float w0 = W[row * 3 + 0], w1 = W[row * 3 + 1], w2 = W[row * 3 + 2];
    float wp    = w0 * pPM[row] + w1 * pPP[row] + w2 * pPF[row];
    float denom = w0 * pPM[row] + w1 * (pPP[row] + nPP[row]) + w2 * (pPF[row] + nPF[row]);
    float nc = fmaxf(dPM[row] + dPP[row] + dPF[row], 1.f);
    float l = wp / fmaxf(denom, 1e-10f) / nc;
    l = fmaxf(l, 1e-10f);
    red[threadIdx.x] = -logf(l);
    __syncthreads();
    for (int s = 128; s; s >>= 1) {
        if (threadIdx.x < s) red[threadIdx.x] += red[threadIdx.x + s];
        __syncthreads();
    }
    if (threadIdx.x == 0) partials[blockIdx.x] = red[0];
}

__global__ void finalize_kernel(const float* __restrict__ partials, float* __restrict__ out) {
    if (threadIdx.x == 0) {
        float s = 0.f;
        for (int i = 0; i < 16; ++i) s += partials[i];
        out[0] = s * (1.f / (float)NNODES);
    }
}

// ---------------------------------------------------------------------------
// launch
// ---------------------------------------------------------------------------
extern "C" void kernel_launch(void* const* d_in, const int* in_sizes, int n_in,
                              void* d_out, int out_size, void* d_ws, size_t ws_size,
                              hipStream_t stream) {
    (void)in_sizes; (void)n_in; (void)out_size; (void)ws_size;
    const float* embP   = (const float*)d_in[0];
    const float* embM   = (const float*)d_in[1];
    const float* embF   = (const float*)d_in[2];
    const float* PM_adj = (const float*)d_in[3];
    const float* PP_adj = (const float*)d_in[4];
    const float* PF_adj = (const float*)d_in[5];
    const float* W1     = (const float*)d_in[6];
    const float* b1     = (const float*)d_in[7];
    const float* W2     = (const float*)d_in[8];
    const float* b2     = (const float*)d_in[9];
    float* out = (float*)d_out;
    float* weights = out + 1;   // output order: (loss, weights[4096,3])

    // workspace carve-out
    char* base = (char*)d_ws;
    size_t off = 0;
    auto take = [&](size_t bytes) -> void* {
        void* p = base + off;
        off = (off + bytes + 255) & ~(size_t)255;
        return p;
    };
    __bf16* Pn = (__bf16*)take((size_t)NNODES * HDIM * 2);
    __bf16* Mn = (__bf16*)take((size_t)NNODES * HDIM * 2);
    __bf16* Fn = (__bf16*)take((size_t)NNODES * HDIM * 2);
    float* feat = (float*)take((size_t)NNODES * 3 * HDIM * 4);
    float* hbuf = (float*)take((size_t)NNODES * HDIM * 4);
    float* degs = (float*)take((size_t)3 * NNODES * 4);   // PM, PP, PF
    float* poss = (float*)take((size_t)3 * NNODES * 4);   // PM, PP, PF (atomics)
    float* negPP = (float*)take((size_t)NNODES * 4);
    float* negPF = (float*)take((size_t)NNODES * 4);
    _Float16* Spp = (_Float16*)take((size_t)NNODES * NNODES * 2);
    _Float16* Spf = (_Float16*)take((size_t)NNODES * NNODES * 2);
    float* partials = (float*)take(64 * 4);
    float *degPM = degs, *degPP = degs + NNODES, *degPF = degs + 2 * NNODES;
    float *posPM = poss, *posPP = poss + NNODES, *posPF = poss + 2 * NNODES;

    zero_kernel<<<(3 * NNODES + 255) / 256, 256, 0, stream>>>(poss, 3 * NNODES);
    prep_norm_kernel<<<NNODES, 256, 0, stream>>>(embP, embM, embF, Pn, Mn, Fn);
    rowsum_kernel<<<dim3(NNODES, 3), 256, 0, stream>>>(PM_adj, PP_adj, PF_adj,
                                                       degPM, degPP, degPF);
    // aggregations: feat = [adj@embM/deg | adj@embP/deg | adj@embF/deg]
    dim3 gAgg(HDIM / 64, NNODES / 64);
    gemm_bf16_kernel<<<gAgg, 256, 0, stream>>>(PM_adj, embM, NNODES, HDIM,
                                               feat, 3 * HDIM, 0,        degPM, nullptr, 0);
    gemm_bf16_kernel<<<gAgg, 256, 0, stream>>>(PP_adj, embP, NNODES, HDIM,
                                               feat, 3 * HDIM, HDIM,     degPP, nullptr, 0);
    gemm_bf16_kernel<<<gAgg, 256, 0, stream>>>(PF_adj, embF, NNODES, HDIM,
                                               feat, 3 * HDIM, 2 * HDIM, degPF, nullptr, 0);
    // MLP layer 1: h = relu(feat @ W1 + b1)
    gemm_bf16_kernel<<<gAgg, 256, 0, stream>>>(feat, W1, 3 * HDIM, HDIM,
                                               hbuf, HDIM, 0, nullptr, b1, 1);
    // layer 2 + double softmax -> weights (directly into d_out)
    mlp2_weights_kernel<<<NNODES / 8, 256, 0, stream>>>(hbuf, W2, b2, weights);
    // similarity GEMMs with fused epilogues
    dim3 gSim(NNODES / 64, NNODES / 64);
    sim_pos_kernel<<<gSim, 256, 0, stream>>>(Pn, Mn, PM_adj, posPM, (_Float16*)nullptr);
    sim_pos_kernel<<<gSim, 256, 0, stream>>>(Pn, Pn, PP_adj, posPP, Spp);
    sim_pos_kernel<<<gSim, 256, 0, stream>>>(Pn, Fn, PF_adj, posPF, Spf);
    // hard-negative selection
    neg_select_kernel<<<NNODES, 256, 0, stream>>>(Spp, degPP, negPP);
    neg_select_kernel<<<NNODES, 256, 0, stream>>>(Spf, degPF, negPF);
    // loss
    final_row_kernel<<<NNODES / 256, 256, 0, stream>>>(weights, posPM, posPP, posPF,
                                                       negPP, negPF, degPM, degPP, degPF,
                                                       partials);
    finalize_kernel<<<1, 32, 0, stream>>>(partials, out);
}
